// Longformer_Local_Attention_31868657336557
// MI455X (gfx1250) — matrix operand order repaired and checked by reading, not compiled, not run on hardware
//
#include <hip/hip_runtime.h>
#include <hip/hip_bf16.h>
#include <cstdint>
#include <cstddef>

#define B_    2
#define S_    4096
#define D_    768
#define H_    12
#define DH_   64
#define NC_   32
#define DFF_  3072
#define MROWS (B_ * S_)          /* 8192 */
#define SCALE_ 0.125f            /* 1/sqrt(64) */
#define LN_EPS 1e-12f

typedef __attribute__((ext_vector_type(16))) _Float16 v16h;
typedef __attribute__((ext_vector_type(8)))  float    v8f;
typedef __attribute__((ext_vector_type(8)))  _Float16 h8;
typedef __attribute__((ext_vector_type(4)))  _Float16 h4;
typedef __attribute__((ext_vector_type(4)))  int      i4;

#define AS1_ __attribute__((address_space(1)))
#define AS3_ __attribute__((address_space(3)))

#if __has_builtin(__builtin_amdgcn_global_load_async_to_lds_b128)
#define USE_ASYNC_COPY 1
#else
#define USE_ASYNC_COPY 0
#endif

#if USE_ASYNC_COPY
// 16-byte global -> LDS async copy (tracked by ASYNCcnt; no VGPR round-trip).
// Builtin signature (from diagnostics): (int4 AS1*, int4 AS3*, imm offset, imm cpol).
__device__ __forceinline__ void async_copy16(const _Float16* g, _Float16* l) {
  __builtin_amdgcn_global_load_async_to_lds_b128((AS1_ i4*)(void*)g,
                                                 (AS3_ i4*)(void*)l, 0, 0);
}
#endif

__device__ __forceinline__ void wait_async0() {
#if __has_builtin(__builtin_amdgcn_s_wait_asynccnt)
  __builtin_amdgcn_s_wait_asynccnt(0);
#else
  asm volatile("s_wait_asynccnt 0" ::: "memory");
#endif
}

// ---------------------------------------------------------------------------
// WMMA fragment helpers (CDNA5 16-bit layouts, wave32).
// A 16x32 (MxK): lane holds row m=lane&15; k pairs at 8*(lane>>4)+{0..3,4..7,16..19,20..23}
// B 32x16 (KxN): lane holds col n=lane&15; k = 8*(lane>>4)+{0..7,16..23}
// ---------------------------------------------------------------------------
__device__ __forceinline__ v16h load_afrag(const _Float16* p) {
  h4 x0 = *(const h4*)(p + 0);
  h4 x1 = *(const h4*)(p + 4);
  h4 x2 = *(const h4*)(p + 16);
  h4 x3 = *(const h4*)(p + 20);
  v16h a;
  a[0]  = x0[0]; a[1]  = x0[1]; a[2]  = x0[2]; a[3]  = x0[3];
  a[4]  = x1[0]; a[5]  = x1[1]; a[6]  = x1[2]; a[7]  = x1[3];
  a[8]  = x2[0]; a[9]  = x2[1]; a[10] = x2[2]; a[11] = x2[3];
  a[12] = x3[0]; a[13] = x3[1]; a[14] = x3[2]; a[15] = x3[3];
  return a;
}

__device__ __forceinline__ v16h load_bfrag(const _Float16* p) {
  h8 y0 = *(const h8*)(p + 0);
  h8 y1 = *(const h8*)(p + 16);
  v16h b;
#pragma unroll
  for (int i = 0; i < 8; ++i) { b[i] = y0[i]; b[8 + i] = y1[i]; }
  return b;
}

__device__ __forceinline__ v8f wmma_f16(v16h a, v16h b, v8f c) {
  return __builtin_amdgcn_wmma_f32_16x16x32_f16(false, a, false, b, (short)0, c,
                                                false, false);
}

// ---------------------------------------------------------------------------
// Weight convert + transpose: W[K][N] f32 -> Wt[N][K] f16
// ---------------------------------------------------------------------------
__global__ __launch_bounds__(256) void wtrans_kernel(const float* __restrict__ W,
                                                     _Float16* __restrict__ Wt,
                                                     int K, int N) {
  int idx = blockIdx.x * 256 + threadIdx.x;
  if (idx < K * N) {
    int k = idx / N, n = idx - k * N;
    Wt[(size_t)n * K + k] = (_Float16)W[idx];
  }
}

// ---------------------------------------------------------------------------
// Row LayerNorm helpers: one block (256 threads) per row of 768
// ---------------------------------------------------------------------------
__device__ __forceinline__ float block_sum(float v, float* red, int tid) {
  red[tid] = v;
  __syncthreads();
#pragma unroll
  for (int s = 128; s > 0; s >>= 1) {
    if (tid < s) red[tid] += red[tid + s];
    __syncthreads();
  }
  float r = red[0];
  __syncthreads();
  return r;
}

__global__ __launch_bounds__(256) void embed_ln_kernel(
    const float* __restrict__ inp, const float* __restrict__ pos,
    const float* __restrict__ tok, const float* __restrict__ g,
    const float* __restrict__ bb, float* __restrict__ xo,
    _Float16* __restrict__ xh) {
  __shared__ float red[256];
  int r = blockIdx.x, tid = threadIdx.x;
  int s = r % S_;
  size_t base = (size_t)r * D_;
  float vals[3];
#pragma unroll
  for (int i = 0; i < 3; ++i) {
    int d = tid + i * 256;
    vals[i] = inp[base + d] + pos[(size_t)(s + 2) * D_ + d] + tok[d];
  }
  float mean = block_sum(vals[0] + vals[1] + vals[2], red, tid) * (1.0f / D_);
  float vs = 0.0f;
#pragma unroll
  for (int i = 0; i < 3; ++i) { float d = vals[i] - mean; vs += d * d; }
  float var = block_sum(vs, red, tid) * (1.0f / D_);
  float inv = rsqrtf(var + LN_EPS);
#pragma unroll
  for (int i = 0; i < 3; ++i) {
    int d = tid + i * 256;
    float o = (vals[i] - mean) * inv * g[d] + bb[d];
    xo[base + d] = o;
    xh[base + d] = (_Float16)o;
  }
}

__global__ __launch_bounds__(256) void add_ln_kernel(
    const float* __restrict__ xin, const float* __restrict__ delta,
    const float* __restrict__ g, const float* __restrict__ bb,
    float* __restrict__ xo, _Float16* __restrict__ xh) {
  __shared__ float red[256];
  int r = blockIdx.x, tid = threadIdx.x;
  size_t base = (size_t)r * D_;
  float vals[3];
#pragma unroll
  for (int i = 0; i < 3; ++i) {
    int d = tid + i * 256;
    vals[i] = xin[base + d] + delta[base + d];
  }
  float mean = block_sum(vals[0] + vals[1] + vals[2], red, tid) * (1.0f / D_);
  float vs = 0.0f;
#pragma unroll
  for (int i = 0; i < 3; ++i) { float d = vals[i] - mean; vs += d * d; }
  float var = block_sum(vs, red, tid) * (1.0f / D_);
  float inv = rsqrtf(var + LN_EPS);
#pragma unroll
  for (int i = 0; i < 3; ++i) {
    int d = tid + i * 256;
    float o = (vals[i] - mean) * inv * g[d] + bb[d];
    xo[base + d] = o;
    xh[base + d] = (_Float16)o;
  }
}

// ---------------------------------------------------------------------------
// GEMM: C[M][N] = A[M][K](f16) @ Bt[N][K](f16)^T + bias, optional exact GELU.
// 128x128 tile, TK=32; 8 waves, each wave = 64x32 region = 4x2 WMMA subtiles.
// Tiles staged into LDS via async global->LDS copies (ASYNCcnt) when available.
// ---------------------------------------------------------------------------
__global__ __launch_bounds__(256) void gemm_kernel(
    const _Float16* __restrict__ A, const _Float16* __restrict__ Bt,
    const float* __restrict__ bias, float* __restrict__ Cf,
    _Float16* __restrict__ Ch, int M, int N, int K, int act) {
  __shared__ _Float16 As[128 * 32];
  __shared__ _Float16 Bs[128 * 32];
  int tid = threadIdx.x;
  int lane = tid & 31, wave = tid >> 5;
  int half = lane >> 4, ln = lane & 15;
  int wm = (wave >> 2) * 64;  // 0 or 64
  int wn = (wave & 3) * 32;   // 0..96
  long m0 = (long)blockIdx.y * 128;
  long n0 = (long)blockIdx.x * 128;

  v8f acc[4][2] = {};

  for (int k0 = 0; k0 < K; k0 += 32) {
    __syncthreads();
#pragma unroll
    for (int it = 0; it < 2; ++it) {
      int idx = tid + it * 256;        // 0..511
      int row = idx >> 2;              // 4 chunks of 8 per 32-wide row
      int col = (idx & 3) * 8;
      const _Float16* ga = A + (m0 + row) * (long)K + k0 + col;
      const _Float16* gb = Bt + (n0 + row) * (long)K + k0 + col;
#if USE_ASYNC_COPY
      async_copy16(ga, As + row * 32 + col);
      async_copy16(gb, Bs + row * 32 + col);
#else
      *(h8*)(As + row * 32 + col) = *(const h8*)ga;
      *(h8*)(Bs + row * 32 + col) = *(const h8*)gb;
#endif
      if (k0 + 32 < K) {  // prefetch next K-tile (global_prefetch_b8)
        __builtin_prefetch((const void*)(ga + 32), 0, 3);
        __builtin_prefetch((const void*)(gb + 32), 0, 3);
      }
    }
#if USE_ASYNC_COPY
    wait_async0();
#endif
    __syncthreads();
    v16h af[4], bfr[2];
#pragma unroll
    for (int i = 0; i < 4; ++i)
      af[i] = load_afrag(As + (wm + i * 16 + ln) * 32 + 8 * half);
#pragma unroll
    for (int j = 0; j < 2; ++j)
      bfr[j] = load_bfrag(Bs + (wn + j * 16 + ln) * 32 + 8 * half);
#pragma unroll
    for (int i = 0; i < 4; ++i)
#pragma unroll
      for (int j = 0; j < 2; ++j)
        acc[i][j] = wmma_f16(af[i], bfr[j], acc[i][j]);
  }

#pragma unroll
  for (int i = 0; i < 4; ++i) {
#pragma unroll
    for (int j = 0; j < 2; ++j) {
      long col = n0 + wn + j * 16 + ln;
      float bv = bias ? bias[col] : 0.0f;
#pragma unroll
      for (int v = 0; v < 8; ++v) {
        long row = m0 + wm + i * 16 + v + 8 * half;
        float val = acc[i][j][v] + bv;
        if (act == 1) val = 0.5f * val * (1.0f + erff(val * 0.70710678118654752f));
        if (Cf) Cf[row * (long)N + col] = val;
        if (Ch) Ch[row * (long)N + col] = (_Float16)val;
      }
    }
  }
}

// ---------------------------------------------------------------------------
// Sliding-window attention. One block per (batch, head, 128-query chunk).
// LDS: Q[128][64], Kwin[384][64], V^T[64][384], probs[128][384] (all f16).
// ---------------------------------------------------------------------------
#define ATT_LDS_ELEMS (128 * 64 + 384 * 64 + 64 * 384 + 128 * 384)

__global__ __launch_bounds__(256) void attn_kernel(
    const _Float16* __restrict__ q, const _Float16* __restrict__ k,
    const _Float16* __restrict__ v, const int* __restrict__ lengths,
    _Float16* __restrict__ ctxh) {
  extern __shared__ _Float16 smem[];
  _Float16* Qh = smem;              // [128][64]
  _Float16* Kh = Qh + 128 * 64;     // [384][64]
  _Float16* Vt = Kh + 384 * 64;     // [64][384]
  _Float16* Pr = Vt + 64 * 384;     // [128][384]

  int blk = blockIdx.x;
  int c  = blk & 31;                // NC=32
  int hh = (blk >> 5) % H_;
  int b  = blk / (32 * H_);
  int len = lengths[b];
  int tid = threadIdx.x;
  int lane = tid & 31, wave = tid >> 5;
  int half = lane >> 4, ln = lane & 15;
  long rowbase = (long)b * S_ + (long)c * 128;

  // Stage Q (pre-scaled by 1/sqrt(dh)).
  for (int t = tid * 4; t < 128 * 64; t += 1024) {
    int i = t >> 6, d = t & 63;
    h4 val = *(const h4*)(q + (rowbase + i) * D_ + hh * DH_ + d);
    h4 o;
#pragma unroll
    for (int jj = 0; jj < 4; ++jj) o[jj] = (_Float16)((float)val[jj] * SCALE_);
    *(h4*)(Qh + i * 64 + d) = o;
  }
  // Stage K window (verbatim f16 copy -> async path; OOB rows zero-filled).
  for (int t = tid * 8; t < 384 * 64; t += 2048) {
    int j = t >> 6, d = t & 63;
    int s = c * 128 - 128 + j;
    if (s >= 0 && s < S_) {
#if USE_ASYNC_COPY
      async_copy16(k + ((long)b * S_ + s) * D_ + hh * DH_ + d, Kh + j * 64 + d);
#else
      *(h8*)(Kh + j * 64 + d) =
          *(const h8*)(k + ((long)b * S_ + s) * D_ + hh * DH_ + d);
#endif
    } else {
      h8 z = {};
      *(h8*)(Kh + j * 64 + d) = z;
    }
  }
  // Stage V^T window (transposed in LDS; OOB rows zero-filled).
  for (int t = tid * 4; t < 384 * 64; t += 1024) {
    int j = t >> 6, d = t & 63;
    int s = c * 128 - 128 + j;
    h4 vv = {};
    if (s >= 0 && s < S_)
      vv = *(const h4*)(v + ((long)b * S_ + s) * D_ + hh * DH_ + d);
#pragma unroll
    for (int jj = 0; jj < 4; ++jj) Vt[(d + jj) * 384 + j] = vv[jj];
  }
#if USE_ASYNC_COPY
  wait_async0();
#endif
  __syncthreads();

  int qs = wave * 16;  // 16-query strip per wave

  // Pass 1: scores = Q K^T with band + length mask, stored f16 into Pr.
  v16h aq0 = load_afrag(Qh + (qs + ln) * 64 + 8 * half);
  v16h aq1 = load_afrag(Qh + (qs + ln) * 64 + 32 + 8 * half);
  for (int kt = 0; kt < 24; ++kt) {
    v8f acc = {};
    v16h b0 = load_bfrag(Kh + (kt * 16 + ln) * 64 + 8 * half);
    acc = wmma_f16(aq0, b0, acc);
    v16h b1 = load_bfrag(Kh + (kt * 16 + ln) * 64 + 32 + 8 * half);
    acc = wmma_f16(aq1, b1, acc);
    int n = kt * 16 + ln;
    int jg = c * 128 + n - 128;
    bool kvalid = (jg >= 0) && (jg < len);
#pragma unroll
    for (int vv2 = 0; vv2 < 8; ++vv2) {
      int m = qs + vv2 + 8 * half;
      bool ok = kvalid && (n >= m) && (n <= m + 256);
      float sv = ok ? acc[vv2] : -60000.0f;
      Pr[m * 384 + n] = (_Float16)sv;
    }
  }
  __syncthreads();

  // Pass 2: row softmax. Lane handles row qs+ln; halves split 384 cols.
  {
    int m = qs + ln;
    _Float16* prow = Pr + m * 384 + half * 192;
    float mx = -1e30f;
    for (int j2 = 0; j2 < 192; j2 += 4) {
      h4 t = *(const h4*)(prow + j2);
#pragma unroll
      for (int jj = 0; jj < 4; ++jj) mx = fmaxf(mx, (float)t[jj]);
    }
    mx = fmaxf(mx, __shfl_xor(mx, 16, 32));
    float sum = 0.0f;
    for (int j2 = 0; j2 < 192; j2 += 4) {
      h4 t = *(const h4*)(prow + j2);
      h4 o;
#pragma unroll
      for (int jj = 0; jj < 4; ++jj) {
        float p = __expf((float)t[jj] - mx);
        sum += p;
        o[jj] = (_Float16)p;
      }
      *(h4*)(prow + j2) = o;
    }
    sum += __shfl_xor(sum, 16, 32);
    float scl = ((c * 128 + m) < len) ? (1.0f / sum) : 0.0f;  // HF query mask
    for (int j2 = 0; j2 < 192; j2 += 4) {
      h4 t = *(const h4*)(prow + j2);
      h4 o;
#pragma unroll
      for (int jj = 0; jj < 4; ++jj) o[jj] = (_Float16)((float)t[jj] * scl);
      *(h4*)(prow + j2) = o;
    }
  }
  __syncthreads();

  // Pass 3: ctx = probs @ V  (128x384 @ 384x64), 4 col-subtiles per wave.
  v8f cacc[4] = {};
  for (int ks = 0; ks < 12; ++ks) {
    v16h ap = load_afrag(Pr + (qs + ln) * 384 + ks * 32 + 8 * half);
#pragma unroll
    for (int nt = 0; nt < 4; ++nt) {
      v16h bp = load_bfrag(Vt + (nt * 16 + ln) * 384 + ks * 32 + 8 * half);
      cacc[nt] = wmma_f16(ap, bp, cacc[nt]);
    }
  }
#pragma unroll
  for (int nt = 0; nt < 4; ++nt) {
#pragma unroll
    for (int vv2 = 0; vv2 < 8; ++vv2) {
      long row = rowbase + qs + vv2 + 8 * half;
      int col = hh * DH_ + nt * 16 + ln;
      ctxh[row * D_ + col] = (_Float16)cacc[nt][vv2];
    }
  }
}

// ---------------------------------------------------------------------------
// Host orchestration
// ---------------------------------------------------------------------------
extern "C" void kernel_launch(void* const* d_in, const int* in_sizes, int n_in,
                              void* d_out, int out_size, void* d_ws,
                              size_t ws_size, hipStream_t stream) {
  (void)in_sizes; (void)n_in; (void)out_size; (void)ws_size;

  const float* inputs   = (const float*)d_in[0];
  const int*   lengths  = (const int*)d_in[1];
  const float* pos_emb  = (const float*)d_in[2];
  const float* tok_emb  = (const float*)d_in[3];
  const float* ln_emb_g = (const float*)d_in[4];
  const float* ln_emb_b = (const float*)d_in[5];
  const float* Wq = (const float*)d_in[6];   const float* bq = (const float*)d_in[7];
  const float* Wk = (const float*)d_in[8];   const float* bk = (const float*)d_in[9];
  const float* Wv = (const float*)d_in[10];  const float* bv = (const float*)d_in[11];
  const float* Wo = (const float*)d_in[12];  const float* bo = (const float*)d_in[13];
  const float* ln1_g = (const float*)d_in[14]; const float* ln1_b = (const float*)d_in[15];
  const float* Wi = (const float*)d_in[16];  const float* bi = (const float*)d_in[17];
  const float* Wf = (const float*)d_in[18];  const float* bf = (const float*)d_in[19];
  const float* ln2_g = (const float*)d_in[20]; const float* ln2_b = (const float*)d_in[21];

  char* p = (char*)d_ws;
  auto take = [&](size_t bytes) {
    char* r = p;
    p += (bytes + 255) & ~(size_t)255;
    return r;
  };
  float*    x    = (float*)take((size_t)MROWS * D_ * 4);
  _Float16* xh   = (_Float16*)take((size_t)MROWS * D_ * 2);
  _Float16* qh   = (_Float16*)take((size_t)MROWS * D_ * 2);
  _Float16* kh   = (_Float16*)take((size_t)MROWS * D_ * 2);
  _Float16* vh   = (_Float16*)take((size_t)MROWS * D_ * 2);
  _Float16* ctxh = (_Float16*)take((size_t)MROWS * D_ * 2);
  float*    yf   = (float*)take((size_t)MROWS * D_ * 4);
  _Float16* h1h  = (_Float16*)take((size_t)MROWS * DFF_ * 2);
  _Float16* WqT  = (_Float16*)take((size_t)D_ * D_ * 2);
  _Float16* WkT  = (_Float16*)take((size_t)D_ * D_ * 2);
  _Float16* WvT  = (_Float16*)take((size_t)D_ * D_ * 2);
  _Float16* WoT  = (_Float16*)take((size_t)D_ * D_ * 2);
  _Float16* WiT  = (_Float16*)take((size_t)D_ * DFF_ * 2);
  _Float16* WfT  = (_Float16*)take((size_t)D_ * DFF_ * 2);

  (void)hipFuncSetAttribute((const void*)attn_kernel,
                            hipFuncAttributeMaxDynamicSharedMemorySize,
                            (int)(ATT_LDS_ELEMS * sizeof(_Float16)));

  embed_ln_kernel<<<MROWS, 256, 0, stream>>>(inputs, pos_emb, tok_emb, ln_emb_g,
                                             ln_emb_b, x, xh);

  dim3 g768(D_ / 128, MROWS / 128);     // (6, 64)
  dim3 g3072(DFF_ / 128, MROWS / 128);  // (24, 64)
  int tb1 = (D_ * D_ + 255) / 256;
  int tb2 = (D_ * DFF_ + 255) / 256;

  for (int l = 0; l < 4; ++l) {
    wtrans_kernel<<<tb1, 256, 0, stream>>>(Wq + (size_t)l * D_ * D_, WqT, D_, D_);
    wtrans_kernel<<<tb1, 256, 0, stream>>>(Wk + (size_t)l * D_ * D_, WkT, D_, D_);
    wtrans_kernel<<<tb1, 256, 0, stream>>>(Wv + (size_t)l * D_ * D_, WvT, D_, D_);
    wtrans_kernel<<<tb1, 256, 0, stream>>>(Wo + (size_t)l * D_ * D_, WoT, D_, D_);
    wtrans_kernel<<<tb2, 256, 0, stream>>>(Wi + (size_t)l * D_ * DFF_, WiT, D_, DFF_);
    wtrans_kernel<<<tb2, 256, 0, stream>>>(Wf + (size_t)l * DFF_ * D_, WfT, DFF_, D_);

    gemm_kernel<<<g768, 256, 0, stream>>>(xh, WqT, bq + l * D_, nullptr, qh,
                                          MROWS, D_, D_, 0);
    gemm_kernel<<<g768, 256, 0, stream>>>(xh, WkT, bk + l * D_, nullptr, kh,
                                          MROWS, D_, D_, 0);
    gemm_kernel<<<g768, 256, 0, stream>>>(xh, WvT, bv + l * D_, nullptr, vh,
                                          MROWS, D_, D_, 0);

    attn_kernel<<<B_ * H_ * NC_, 256, ATT_LDS_ELEMS * sizeof(_Float16), stream>>>(
        qh, kh, vh, lengths, ctxh);

    gemm_kernel<<<g768, 256, 0, stream>>>(ctxh, WoT, bo + l * D_, yf, nullptr,
                                          MROWS, D_, D_, 0);
    add_ln_kernel<<<MROWS, 256, 0, stream>>>(x, yf, ln1_g + l * D_,
                                             ln1_b + l * D_, x, xh);

    gemm_kernel<<<g3072, 256, 0, stream>>>(xh, WiT, bi + l * DFF_, nullptr, h1h,
                                           MROWS, DFF_, D_, 1);
    gemm_kernel<<<g768, 256, 0, stream>>>(h1h, WfT, bf + l * D_, yf, nullptr,
                                          MROWS, D_, DFF_, 0);
    float* xout = (l == 3) ? (float*)d_out : x;
    add_ln_kernel<<<MROWS, 256, 0, stream>>>(x, yf, ln2_g + l * D_,
                                             ln2_b + l * D_, xout, xh);
  }
}
